// SLRDifferentialModule_9234179686574
// MI455X (gfx1250) — matrix operand order repaired and checked by reading, not compiled, not run on hardware
//
#include <hip/hip_runtime.h>
#include <hip/hip_bf16.h>

typedef _Float16 half_t;
typedef __attribute__((ext_vector_type(16))) _Float16 v16h;
typedef __attribute__((ext_vector_type(8)))  _Float16 v8h;
typedef __attribute__((ext_vector_type(8)))  float    v8f;

#define WMMA16(a, b, c) \
  __builtin_amdgcn_wmma_f32_16x16x32_f16(false, (a), false, (b), (short)0, (c), false, false)

__device__ __forceinline__ v8f zero8() {
  v8f z;
#pragma unroll
  for (int i = 0; i < 8; ++i) z[i] = 0.f;
  return z;
}

// A fragment (16x32 f16): rowbase points at (row=lane&15, k0) in a row-major
// source. Halves 0-7 -> K = hi*8 + h ; halves 8-15 -> K = hi*8 + 16 + h.
__device__ __forceinline__ v16h load_a_f16(const half_t* rowbase, int lhi) {
  const half_t* p = rowbase + lhi * 8;
  v8h lo = *(const v8h*)p;
  v8h hi = *(const v8h*)(p + 16);
  v16h a;
#pragma unroll
  for (int j = 0; j < 8; ++j) { a[j] = lo[j]; a[8 + j] = hi[j]; }
  return a;
}

// A fragment from f32 row-major source, converting to f16 on the fly.
__device__ __forceinline__ v16h load_a_f32(const float* rowbase, int lhi) {
  const float* p = rowbase + lhi * 8;
  const float4* q0 = (const float4*)p;
  const float4* q1 = (const float4*)(p + 16);
  float4 u0 = q0[0], u1 = q0[1], u2 = q1[0], u3 = q1[1];
  v16h a;
  a[0] = (_Float16)u0.x;  a[1] = (_Float16)u0.y;  a[2] = (_Float16)u0.z;  a[3] = (_Float16)u0.w;
  a[4] = (_Float16)u1.x;  a[5] = (_Float16)u1.y;  a[6] = (_Float16)u1.z;  a[7] = (_Float16)u1.w;
  a[8] = (_Float16)u2.x;  a[9] = (_Float16)u2.y;  a[10] = (_Float16)u2.z; a[11] = (_Float16)u2.w;
  a[12] = (_Float16)u3.x; a[13] = (_Float16)u3.y; a[14] = (_Float16)u3.z; a[15] = (_Float16)u3.w;
  return a;
}

// B fragment (32x16 f16): element (K=k, N=n) taken from base[n*ld + k]
// (i.e. contiguous K-runs per output column). lane&15 = N, halves 0-7 -> K = hi*8 + h,
// halves 8-15 -> K = hi*8 + 16 + h.
__device__ __forceinline__ v16h load_b_f16(const half_t* base, int ld, int lane) {
  const int n = lane & 15, lhi = lane >> 4;
  const half_t* p = base + (size_t)n * ld + lhi * 8;
  v8h lo = *(const v8h*)p;
  v8h hi = *(const v8h*)(p + 16);
  v16h b;
#pragma unroll
  for (int j = 0; j < 8; ++j) { b[j] = lo[j]; b[8 + j] = hi[j]; }
  return b;
}

// ---------------------------------------------------------------------------
// Kernel 0: convert weights to f16
// ---------------------------------------------------------------------------
__global__ void cvt_weights_kernel(const float* __restrict__ wq,
                                   const float* __restrict__ wkv,
                                   const float* __restrict__ wout,
                                   half_t* __restrict__ wq16,
                                   half_t* __restrict__ wkv16,
                                   half_t* __restrict__ wout16) {
  int i = blockIdx.x * blockDim.x + threadIdx.x;
  if (i < 131072) wq16[i] = (_Float16)wq[i];     // 128*1024
  if (i < 196608) wkv16[i] = (_Float16)wkv[i];   // 192*1024
  if (i < 65536)  wout16[i] = (_Float16)wout[i]; // 1024*64
}

// ---------------------------------------------------------------------------
// Kernel 1: per-batch top-1024 of 8192 via bitonic sort in LDS (descending)
// ---------------------------------------------------------------------------
__global__ __launch_bounds__(1024) void topk_kernel(const float* __restrict__ imp,
                                                    int* __restrict__ top_idx) {
  __shared__ unsigned skey[8192];
  __shared__ unsigned short sidx[8192];
  const int b = blockIdx.x, tid = threadIdx.x;
  for (int i = tid; i < 8192; i += 1024) {
    unsigned u = __float_as_uint(imp[b * 8192 + i]);
    u = (u & 0x80000000u) ? ~u : (u | 0x80000000u); // order-preserving key
    skey[i] = u;
    sidx[i] = (unsigned short)i;
  }
  __syncthreads();
  for (int k = 2; k <= 8192; k <<= 1) {
    for (int j = k >> 1; j > 0; j >>= 1) {
      for (int i = tid; i < 8192; i += 1024) {
        int ixj = i ^ j;
        if (ixj > i) {
          unsigned ki = skey[i], kj = skey[ixj];
          bool up = (i & k) == 0;
          bool sw = up ? (ki < kj) : (ki > kj); // descending overall
          if (sw) {
            skey[i] = kj; skey[ixj] = ki;
            unsigned short t = sidx[i]; sidx[i] = sidx[ixj]; sidx[ixj] = t;
          }
        }
      }
      __syncthreads();
    }
  }
  if (tid < 1024) top_idx[b * 1024 + tid] = (int)sidx[tid];
}

// ---------------------------------------------------------------------------
// Kernel 2: kv = x @ W_kv^T, split-stored as K1[tok][64], K2[tok][64],
//           Vt[b][dh][s] (transposed for attention B-fragments).
// grid 512 x 256 threads: wave -> 16 rows x 192 cols, K-loop 1024.
// ---------------------------------------------------------------------------
__global__ __launch_bounds__(256) void kv_proj_kernel(const float* __restrict__ x,
                                                      const half_t* __restrict__ wkv,
                                                      half_t* __restrict__ k1g,
                                                      half_t* __restrict__ k2g,
                                                      half_t* __restrict__ vtg) {
  const int tid = threadIdx.x, wave = tid >> 5, lane = tid & 31;
  const int lhi = lane >> 4, ln = lane & 15;
  const int row0 = blockIdx.x * 128 + wave * 16; // token row in [0, 65536)
  const float* xrow = x + (size_t)(row0 + ln) * 1024;

  v8f acc[12];
#pragma unroll
  for (int t = 0; t < 12; ++t) acc[t] = zero8();

  for (int ks = 0; ks < 1024; ks += 32) {
    if (ks + 32 < 1024) __builtin_prefetch(xrow + ks + 32, 0, 1);
    v16h a = load_a_f32(xrow + ks, lhi);
#pragma unroll
    for (int t = 0; t < 12; ++t) {
      v16h bw = load_b_f16(wkv + (size_t)(t * 16) * 1024 + ks, 1024, lane);
      acc[t] = WMMA16(a, bw, acc[t]);
    }
  }
#pragma unroll
  for (int t = 0; t < 12; ++t) {
    const int e0 = t * 16, e = e0 + ln;
#pragma unroll
    for (int r = 0; r < 8; ++r) {
      const int row = row0 + r + lhi * 8;
      const float v = acc[t][r];
      if (e0 < 64) {
        k1g[(size_t)row * 64 + e] = (_Float16)v;
      } else if (e0 < 128) {
        k2g[(size_t)row * 64 + (e - 64)] = (_Float16)v;
      } else {
        const int bb = row >> 13, ss = row & 8191;
        vtg[(size_t)bb * 64 * 8192 + (size_t)(e - 128) * 8192 + ss] = (_Float16)v;
      }
    }
  }
}

// ---------------------------------------------------------------------------
// Kernel 3: q = gather(x, top_idx) @ W_q^T -> Q1[qrow][64], Q2[qrow][64]
// grid 64 x 256: wave -> 16 gathered rows x 128 cols.
// ---------------------------------------------------------------------------
__global__ __launch_bounds__(256) void q_proj_kernel(const float* __restrict__ x,
                                                     const half_t* __restrict__ wq,
                                                     const int* __restrict__ top_idx,
                                                     half_t* __restrict__ q1,
                                                     half_t* __restrict__ q2) {
  const int tid = threadIdx.x, wave = tid >> 5, lane = tid & 31;
  const int lhi = lane >> 4, ln = lane & 15;
  const int qrow0 = blockIdx.x * 128 + wave * 16; // in [0, 8192)
  const int b = qrow0 >> 10;
  const int src = top_idx[qrow0 + ln];
  const float* xrow = x + ((size_t)b * 8192 + src) * 1024;

  v8f acc[8];
#pragma unroll
  for (int t = 0; t < 8; ++t) acc[t] = zero8();

  for (int ks = 0; ks < 1024; ks += 32) {
    v16h a = load_a_f32(xrow + ks, lhi);
#pragma unroll
    for (int t = 0; t < 8; ++t) {
      v16h bw = load_b_f16(wq + (size_t)(t * 16) * 1024 + ks, 1024, lane);
      acc[t] = WMMA16(a, bw, acc[t]);
    }
  }
#pragma unroll
  for (int t = 0; t < 8; ++t) {
    const int e = t * 16 + ln;
#pragma unroll
    for (int r = 0; r < 8; ++r) {
      const int row = qrow0 + r + lhi * 8;
      const float v = acc[t][r];
      if (e < 64) q1[(size_t)row * 64 + e] = (_Float16)v;
      else        q2[(size_t)row * 64 + (e - 64)] = (_Float16)v;
    }
  }
}

// ---------------------------------------------------------------------------
// Kernel 4: dual-softmax flash attention + RMS norm -> h16[qrow][64]
// grid 512 (= 8 batches x 64 query tiles of 16), 8 waves split S=8192.
// ---------------------------------------------------------------------------
__global__ __launch_bounds__(256) void attn_kernel(const half_t* __restrict__ q1,
                                                   const half_t* __restrict__ q2,
                                                   const half_t* __restrict__ k1g,
                                                   const half_t* __restrict__ k2g,
                                                   const half_t* __restrict__ vtg,
                                                   const float* __restrict__ lam_logit,
                                                   const float* __restrict__ rms_weight,
                                                   half_t* __restrict__ h16) {
  __shared__ __align__(32) half_t P1buf[8][16 * 32];
  __shared__ __align__(32) half_t P2buf[8][16 * 32];
  __shared__ float mls[8][2][16];
  __shared__ float accS[2][16][64];
  __shared__ float lS[2][16];

  const int tid = threadIdx.x, wave = tid >> 5, lane = tid & 31;
  const int lhi = lane >> 4, ln = lane & 15;
  const int qt = blockIdx.x;
  const int b = qt >> 6;
  const int qrow0 = qt * 16;

  for (int i = tid; i < 2 * 16 * 64; i += 256) ((float*)accS)[i] = 0.f;
  for (int i = tid; i < 2 * 16; i += 256) ((float*)lS)[i] = 0.f;
  __syncthreads();

  const half_t* qb1 = q1 + (size_t)(qrow0 + ln) * 64;
  const half_t* qb2 = q2 + (size_t)(qrow0 + ln) * 64;
  v16h a1lo = load_a_f16(qb1 + 0, lhi);
  v16h a1hi = load_a_f16(qb1 + 32, lhi);
  v16h a2lo = load_a_f16(qb2 + 0, lhi);
  v16h a2hi = load_a_f16(qb2 + 32, lhi);

  v8f acc1[4], acc2[4];
#pragma unroll
  for (int t = 0; t < 4; ++t) { acc1[t] = zero8(); acc2[t] = zero8(); }
  float m1[8], l1[8], m2[8], l2[8];
#pragma unroll
  for (int r = 0; r < 8; ++r) { m1[r] = -1e30f; m2[r] = -1e30f; l1[r] = 0.f; l2[r] = 0.f; }

  const float sc = 0.125f; // 1/sqrt(64)
  const half_t* k1b = k1g + (size_t)b * 8192 * 64;
  const half_t* k2b = k2g + (size_t)b * 8192 * 64;
  const half_t* vb = vtg + (size_t)b * 64 * 8192;
  half_t* P1 = P1buf[wave];
  half_t* P2 = P2buf[wave];

  for (int c = 0; c < 32; ++c) {
    const int s = wave * 1024 + c * 32;
    // ---- scores: two 16-key subtiles per head, contraction dh=64 (2 ksteps)
    v8f s1a = zero8(), s1b = zero8(), s2a = zero8(), s2b = zero8();
    {
      v16h bk;
      bk = load_b_f16(k1b + (size_t)s * 64 + 0, 64, lane);         s1a = WMMA16(a1lo, bk, s1a);
      bk = load_b_f16(k1b + (size_t)s * 64 + 32, 64, lane);        s1a = WMMA16(a1hi, bk, s1a);
      bk = load_b_f16(k1b + (size_t)(s + 16) * 64 + 0, 64, lane);  s1b = WMMA16(a1lo, bk, s1b);
      bk = load_b_f16(k1b + (size_t)(s + 16) * 64 + 32, 64, lane); s1b = WMMA16(a1hi, bk, s1b);
      bk = load_b_f16(k2b + (size_t)s * 64 + 0, 64, lane);         s2a = WMMA16(a2lo, bk, s2a);
      bk = load_b_f16(k2b + (size_t)s * 64 + 32, 64, lane);        s2a = WMMA16(a2hi, bk, s2a);
      bk = load_b_f16(k2b + (size_t)(s + 16) * 64 + 0, 64, lane);  s2b = WMMA16(a2lo, bk, s2b);
      bk = load_b_f16(k2b + (size_t)(s + 16) * 64 + 32, 64, lane); s2b = WMMA16(a2hi, bk, s2b);
    }
    // ---- streaming softmax update (row reductions across 16-lane halves)
    float p1a[8], p1b[8], p2a[8], p2b[8], f1[8], f2[8];
#pragma unroll
    for (int r = 0; r < 8; ++r) {
      float ea = s1a[r] * sc, eb = s1b[r] * sc;
      float cm = fmaxf(ea, eb);
      cm = fmaxf(cm, __shfl_xor(cm, 1, 32));
      cm = fmaxf(cm, __shfl_xor(cm, 2, 32));
      cm = fmaxf(cm, __shfl_xor(cm, 4, 32));
      cm = fmaxf(cm, __shfl_xor(cm, 8, 32));
      float mn = fmaxf(m1[r], cm);
      float f = __expf(m1[r] - mn);
      float pa = __expf(ea - mn), pb = __expf(eb - mn);
      float rs = pa + pb;
      rs += __shfl_xor(rs, 1, 32);
      rs += __shfl_xor(rs, 2, 32);
      rs += __shfl_xor(rs, 4, 32);
      rs += __shfl_xor(rs, 8, 32);
      l1[r] = l1[r] * f + rs; m1[r] = mn; f1[r] = f; p1a[r] = pa; p1b[r] = pb;

      ea = s2a[r] * sc; eb = s2b[r] * sc;
      cm = fmaxf(ea, eb);
      cm = fmaxf(cm, __shfl_xor(cm, 1, 32));
      cm = fmaxf(cm, __shfl_xor(cm, 2, 32));
      cm = fmaxf(cm, __shfl_xor(cm, 4, 32));
      cm = fmaxf(cm, __shfl_xor(cm, 8, 32));
      mn = fmaxf(m2[r], cm);
      f = __expf(m2[r] - mn);
      pa = __expf(ea - mn); pb = __expf(eb - mn);
      rs = pa + pb;
      rs += __shfl_xor(rs, 1, 32);
      rs += __shfl_xor(rs, 2, 32);
      rs += __shfl_xor(rs, 4, 32);
      rs += __shfl_xor(rs, 8, 32);
      l2[r] = l2[r] * f + rs; m2[r] = mn; f2[r] = f; p2a[r] = pa; p2b[r] = pb;
    }
#pragma unroll
    for (int t = 0; t < 4; ++t)
#pragma unroll
      for (int r = 0; r < 8; ++r) { acc1[t][r] *= f1[r]; acc2[t][r] *= f2[r]; }
    // ---- transpose P (C-layout -> row-major LDS) for A-fragment reuse
#pragma unroll
    for (int r = 0; r < 8; ++r) {
      const int row = r + lhi * 8;
      P1[row * 32 + ln] = (_Float16)p1a[r];
      P1[row * 32 + 16 + ln] = (_Float16)p1b[r];
      P2[row * 32 + ln] = (_Float16)p2a[r];
      P2[row * 32 + 16 + ln] = (_Float16)p2b[r];
    }
    // ---- P @ V : one 32-key kstep, 4 dh tiles, V fragments shared by heads
    v16h aP1 = load_a_f16(P1 + ln * 32, lhi);
    v16h aP2 = load_a_f16(P2 + ln * 32, lhi);
#pragma unroll
    for (int t = 0; t < 4; ++t) {
      v16h bv = load_b_f16(vb + (size_t)(t * 16) * 8192 + s, 8192, lane);
      acc1[t] = WMMA16(aP1, bv, acc1[t]);
      acc2[t] = WMMA16(aP2, bv, acc2[t]);
    }
  }

  // ---- cross-wave merge (flash combine of 8 partials)
  if (ln == 0) {
#pragma unroll
    for (int r = 0; r < 8; ++r) {
      mls[wave][0][r + lhi * 8] = m1[r];
      mls[wave][1][r + lhi * 8] = m2[r];
    }
  }
  __syncthreads();
#pragma unroll
  for (int r = 0; r < 8; ++r) {
    const int row = r + lhi * 8;
    float g1 = -1e30f, g2 = -1e30f;
#pragma unroll
    for (int w = 0; w < 8; ++w) {
      g1 = fmaxf(g1, mls[w][0][row]);
      g2 = fmaxf(g2, mls[w][1][row]);
    }
    const float f1v = __expf(m1[r] - g1);
    const float f2v = __expf(m2[r] - g2);
    l1[r] *= f1v; l2[r] *= f2v;
#pragma unroll
    for (int t = 0; t < 4; ++t) { acc1[t][r] *= f1v; acc2[t][r] *= f2v; }
  }
#pragma unroll
  for (int t = 0; t < 4; ++t)
#pragma unroll
    for (int r = 0; r < 8; ++r) {
      atomicAdd(&accS[0][r + lhi * 8][t * 16 + ln], acc1[t][r]);
      atomicAdd(&accS[1][r + lhi * 8][t * 16 + ln], acc2[t][r]);
    }
  if (ln == 0) {
#pragma unroll
    for (int r = 0; r < 8; ++r) {
      atomicAdd(&lS[0][r + lhi * 8], l1[r]);
      atomicAdd(&lS[1][r + lhi * 8], l2[r]);
    }
  }
  __syncthreads();

  // ---- finalize: attn = A1V - lam*A2V, RMS norm, write h (wave 0 only)
  if (wave == 0) {
    const float lam = 1.f / (1.f + __expf(-lam_logit[0]));
    const int row = lane >> 1;
    const int cb = (lane & 1) * 32;
    const float inv_l1 = 1.f / lS[0][row];
    const float inv_l2 = 1.f / lS[1][row];
    float ss = 0.f;
#pragma unroll
    for (int j = 0; j < 32; ++j) {
      float a = accS[0][row][cb + j] * inv_l1 - lam * accS[1][row][cb + j] * inv_l2;
      ss += a * a;
    }
    ss += __shfl_xor(ss, 1, 32);
    const float rinv = rsqrtf(ss / 64.f + 1e-6f);
#pragma unroll
    for (int j = 0; j < 32; ++j) {
      float a = accS[0][row][cb + j] * inv_l1 - lam * accS[1][row][cb + j] * inv_l2;
      h16[(size_t)(qrow0 + row) * 64 + cb + j] = (_Float16)(a * rinv * rms_weight[cb + j]);
    }
  }
}

// ---------------------------------------------------------------------------
// Kernel 5: out = x (bulk float4 copy)
// ---------------------------------------------------------------------------
__global__ __launch_bounds__(256) void copy_kernel(const float4* __restrict__ src,
                                                   float4* __restrict__ dst, int n) {
  int i = blockIdx.x * blockDim.x + threadIdx.x;
  if (i < n) dst[i] = src[i];
}

// ---------------------------------------------------------------------------
// Kernel 6: y = h @ W_out^T + b_out; merged = g*y + (1-g)*x_sel; scatter.
// grid 512 x 256: block -> 16 rows x 1024 cols, wave -> 128 cols.
// ---------------------------------------------------------------------------
__global__ __launch_bounds__(256) void out_proj_kernel(const float* __restrict__ x,
                                                       const half_t* __restrict__ h16,
                                                       const half_t* __restrict__ wout,
                                                       const float* __restrict__ b_out,
                                                       const float* __restrict__ merge_gate,
                                                       const int* __restrict__ top_idx,
                                                       float* __restrict__ out) {
  const int tid = threadIdx.x, wave = tid >> 5, lane = tid & 31;
  const int lhi = lane >> 4, ln = lane & 15;
  const int qrow0 = blockIdx.x * 16;
  const int b = qrow0 >> 10;
  const int d0w = wave * 128;

  const half_t* hrow = h16 + (size_t)(qrow0 + ln) * 64;
  v16h alo = load_a_f16(hrow + 0, lhi);
  v16h ahi = load_a_f16(hrow + 32, lhi);

  v8f acc[8];
#pragma unroll
  for (int t = 0; t < 8; ++t) {
    acc[t] = zero8();
    const int d0 = d0w + t * 16;
    v16h b0 = load_b_f16(wout + (size_t)d0 * 64 + 0, 64, lane);
    acc[t] = WMMA16(alo, b0, acc[t]);
    v16h b1 = load_b_f16(wout + (size_t)d0 * 64 + 32, 64, lane);
    acc[t] = WMMA16(ahi, b1, acc[t]);
  }

  const float gate = merge_gate[0];
  int srcs[8];
#pragma unroll
  for (int r = 0; r < 8; ++r) srcs[r] = top_idx[qrow0 + r + lhi * 8];

#pragma unroll
  for (int t = 0; t < 8; ++t) {
    const int d = d0w + t * 16 + ln;
    const float bo = b_out[d];
#pragma unroll
    for (int r = 0; r < 8; ++r) {
      const size_t gidx = ((size_t)b * 8192 + srcs[r]) * 1024 + d;
      const float y = acc[t][r] + bo;
      out[gidx] = gate * y + (1.f - gate) * x[gidx];
    }
  }
}

// ---------------------------------------------------------------------------
extern "C" void kernel_launch(void* const* d_in, const int* in_sizes, int n_in,
                              void* d_out, int out_size, void* d_ws, size_t ws_size,
                              hipStream_t stream) {
  const float* x = (const float*)d_in[0];
  const float* importance = (const float*)d_in[1];
  const float* W_q = (const float*)d_in[2];
  const float* W_kv = (const float*)d_in[3];
  const float* lam_logit = (const float*)d_in[4];
  const float* rms_weight = (const float*)d_in[5];
  const float* merge_gate = (const float*)d_in[6];
  const float* W_out = (const float*)d_in[7];
  const float* b_out = (const float*)d_in[8];
  float* out = (float*)d_out;

  char* ws = (char*)d_ws;
  size_t o = 0;
  int* top_idx   = (int*)(ws + o);    o += (size_t)8192 * 4;
  half_t* wq16   = (half_t*)(ws + o); o += (size_t)131072 * 2;
  half_t* wkv16  = (half_t*)(ws + o); o += (size_t)196608 * 2;
  half_t* wout16 = (half_t*)(ws + o); o += (size_t)65536 * 2;
  half_t* q1     = (half_t*)(ws + o); o += (size_t)524288 * 2;
  half_t* q2     = (half_t*)(ws + o); o += (size_t)524288 * 2;
  half_t* k1g    = (half_t*)(ws + o); o += (size_t)4194304 * 2;
  half_t* k2g    = (half_t*)(ws + o); o += (size_t)4194304 * 2;
  half_t* vtg    = (half_t*)(ws + o); o += (size_t)4194304 * 2;
  half_t* h16    = (half_t*)(ws + o); o += (size_t)524288 * 2;

  cvt_weights_kernel<<<768, 256, 0, stream>>>(W_q, W_kv, W_out, wq16, wkv16, wout16);
  topk_kernel<<<8, 1024, 0, stream>>>(importance, top_idx);
  kv_proj_kernel<<<512, 256, 0, stream>>>(x, wkv16, k1g, k2g, vtg);
  q_proj_kernel<<<64, 256, 0, stream>>>(x, wq16, top_idx, q1, q2);
  copy_kernel<<<65536, 256, 0, stream>>>((const float4*)x, (float4*)out, 16777216);
  attn_kernel<<<512, 256, 0, stream>>>(q1, q2, k1g, k2g, vtg, lam_logit, rms_weight, h16);
  out_proj_kernel<<<512, 256, 0, stream>>>(x, h16, wout16, b_out, merge_gate, top_idx, out);
}
